// MSLoss_16071767621696
// MI455X (gfx1250) — compile-verified, matrix operand order checked by reading, not run on hardware
//
#include <hip/hip_runtime.h>
#include <hip/hip_bf16.h>

// ---------------------------------------------------------------------------
// MS-loss (multi-similarity) for B=16384, D=128 on MI455X (gfx1250, wave32).
// sim = normalize(X) @ normalize(X)^T computed twice (two fused passes) with
// v_wmma_f32_16x16x32_f16: input is 8 MB (fits the 192 MB L2) so the problem
// is compute bound; f16 WMMA (16384 FLOP/instr) is 8x the f32 WMMA shape and
// |x|<=1 after normalization keeps f16 numerically safe.
//
// Per tile a wave issues 4 chained WMMAs (~64-128 XDL cycles); the epilogue
// is fully branchless VALU that co-executes underneath (WMMA 16-bit ops are
// TRANS-class on CDNA5):
//  - B-tiles are ping-pong double-buffered with the loop unrolled by 2, so
//    prefetch loads overlap the previous tile's WMMAs WITHOUT the register
//    rotation copies a single rotating buffer forces (stable buffer homes).
//  - min/max accumulate via inline-asm v_min_num/v_max_num: fminf/fmaxf lower
//    to llvm.minnum and drag in ~31 canonicalize ops per tile (IEEE sNaN
//    quieting) that are dead weight -- inputs here are never sNaN.
//  - pass A skips the i==j test: self-sim ~1.0 is the max possible sim, so it
//    can only become min_pos for rows with no other positive, which are
//    invalid rows anyway (matches the reference's sim<1-1e-5 tolerance).
//  - pass B carries no keep-counts: "pos_keep nonempty" <=> min_pos-0.1 <
//    max_neg and "neg_keep nonempty" <=> max_neg+0.1 > min_pos are the SAME
//    inequality, so validity is derived in finalize from min/max alone.
// Symmetry (halving WMMA work) was rejected: transposed-tile stats cannot
// stay register-resident and the ~16M cross-wave atomic merges cost more
// cycles than the saved FLOPs.
// ---------------------------------------------------------------------------

typedef __attribute__((ext_vector_type(16))) _Float16 v16h;
typedef __attribute__((ext_vector_type(8)))  _Float16 v8h;
typedef __attribute__((ext_vector_type(8)))  float    v8f;
typedef __attribute__((ext_vector_type(4)))  float    v4f;

#define N_ROWS   16384
#define DIM      128
#define N_TILES  (N_ROWS / 16)      // 1024 16-row tiles
#define COL_CHUNKS 8                // 8192 waves total for latency hiding
#define TILES_PER_CHUNK (N_TILES / COL_CHUNKS)   // 128 (even)
#define K_ALPHA  50.0f
#define K_BETA   2.0f
#define K_MARGIN 0.5f

// ---- raw v_min_num/v_max_num (no canonicalize; inputs are never sNaN) ------
__device__ __forceinline__ float vminf(float a, float b) {
  float d;
  asm("v_min_num_f32 %0, %1, %2" : "=v"(d) : "v"(a), "v"(b));
  return d;
}
__device__ __forceinline__ float vmaxf(float a, float b) {
  float d;
  asm("v_max_num_f32 %0, %1, %2" : "=v"(d) : "v"(a), "v"(b));
  return d;
}

// ---- order-preserving float<->uint encoding (for atomicMin/Max on floats) --
__device__ __forceinline__ unsigned enc_f32(float f) {
  unsigned u = __float_as_uint(f);
  return (u & 0x80000000u) ? ~u : (u | 0x80000000u);
}
__device__ __forceinline__ float dec_f32(unsigned u) {
  u = (u & 0x80000000u) ? (u & 0x7FFFFFFFu) : ~u;
  return __uint_as_float(u);
}
#define ENC_POS_INF 0xFF800000u   // enc(+inf)
#define ENC_NEG_INF 0x007FFFFFu   // enc(-inf)

// ---- load one 16x32 f16 WMMA operand fragment ------------------------------
// ISA layout (CDNA5 ISA): lanes 0-15 hold K = {0..7, 16..23}, lanes 16-31
// hold K = {8..15, 24..31} for row M = lane&15.  Two 16-byte loads per frag
// (compiler merges adjacent fragments into global_load_b128 clauses).
__device__ __forceinline__ v16h load_frag(const _Float16* __restrict__ X,
                                          int row, int kc, int hi) {
  const _Float16* p = X + row * DIM + kc + hi * 8;
  v8h lo = *(const v8h*)p;          // K = kc + hi*8 + [0..7]
  v8h hh = *(const v8h*)(p + 16);   // K = kc + 16 + hi*8 + [0..7]
  return __builtin_shufflevector(lo, hh, 0,1,2,3,4,5,6,7,8,9,10,11,12,13,14,15);
}

__device__ __forceinline__ v8f wmma_f16(v16h a, v16h b, v8f c) {
  return __builtin_amdgcn_wmma_f32_16x16x32_f16(false, a, false, b,
                                                (short)0, c, false, false);
}

// One 16-column B-tile (K=128) + its target labels; lives in registers.
struct BTile {
  v16h b0, b1, b2, b3;
  int  tj;
};
__device__ __forceinline__ BTile load_tile(const _Float16* __restrict__ Xh,
                                           const int* __restrict__ targets,
                                           int ct, int nsub, int hi) {
  BTile t;
  int j = ct * 16 + nsub;
  t.tj = targets[j];
  t.b0 = load_frag(Xh, j,  0, hi);
  t.b1 = load_frag(Xh, j, 32, hi);
  t.b2 = load_frag(Xh, j, 64, hi);
  t.b3 = load_frag(Xh, j, 96, hi);
  return t;
}

// ---------------------------------------------------------------------------
__global__ void init_kernel(unsigned* __restrict__ minKey,
                            unsigned* __restrict__ maxKey,
                            float* __restrict__ ps, float* __restrict__ ns,
                            float* __restrict__ out) {
  int i = blockIdx.x * blockDim.x + threadIdx.x;
  if (i < N_ROWS) {
    minKey[i] = ENC_POS_INF;   // running min over positives (empty => +inf)
    maxKey[i] = ENC_NEG_INF;   // running max over negatives (empty => -inf)
    ps[i] = 0.f; ns[i] = 0.f;
  }
  if (i == 0) out[0] = 0.f;
}

// One wave per row: L2-normalize and convert to f16.
__global__ void __launch_bounds__(256)
normalize_kernel(const float* __restrict__ X, _Float16* __restrict__ Xh) {
  int wave = (blockIdx.x * blockDim.x + threadIdx.x) >> 5;
  int lane = threadIdx.x & 31;
  if (wave >= N_ROWS) return;
  const float* row = X + (size_t)wave * DIM;
  v4f v = *(const v4f*)(row + lane * 4);
  float ss = v.x * v.x + v.y * v.y + v.z * v.z + v.w * v.w;
  #pragma unroll
  for (int m = 16; m >= 1; m >>= 1) ss += __shfl_xor(ss, m, 32);
  float inv = 1.0f / fmaxf(sqrtf(ss), 1e-12f);
  _Float16* o = Xh + (size_t)wave * DIM + lane * 4;
  o[0] = (_Float16)(v.x * inv); o[1] = (_Float16)(v.y * inv);
  o[2] = (_Float16)(v.z * inv); o[3] = (_Float16)(v.w * inv);
}

// ---------------------------------------------------------------------------
// Pass A: per-row min positive sim / max negative sim.
// Wave = (rowTile, colChunk). A-fragments stay resident across column sweep;
// row stats accumulate in registers, one atomic per row at the end.
__global__ void __launch_bounds__(256)
pass_minmax(const _Float16* __restrict__ Xh, const int* __restrict__ targets,
            unsigned* __restrict__ minKey, unsigned* __restrict__ maxKey) {
  int gwave = (blockIdx.x * blockDim.x + threadIdx.x) >> 5;
  int lane  = threadIdx.x & 31;
  int rowTile = gwave / COL_CHUNKS;
  int chunk   = gwave % COL_CHUNKS;
  if (rowTile >= N_TILES) return;
  int hi   = lane >> 4;
  int nsub = lane & 15;
  int rowBase = rowTile * 16;

  v16h a0 = load_frag(Xh, rowBase + nsub,  0, hi);
  v16h a1 = load_frag(Xh, rowBase + nsub, 32, hi);
  v16h a2 = load_frag(Xh, rowBase + nsub, 64, hi);
  v16h a3 = load_frag(Xh, rowBase + nsub, 96, hi);

  int ti[8];
  #pragma unroll
  for (int r = 0; r < 8; ++r) ti[r] = targets[rowBase + r + 8 * hi];

  float rmin[8], rmax[8];
  #pragma unroll
  for (int r = 0; r < 8; ++r) { rmin[r] = __builtin_inff(); rmax[r] = -__builtin_inff(); }

  const float PINF = __builtin_inff();
  auto mm = [&](const BTile& t) {
    v8f acc = {};
    acc = wmma_f16(a0, t.b0, acc);
    acc = wmma_f16(a1, t.b1, acc);
    acc = wmma_f16(a2, t.b2, acc);
    acc = wmma_f16(a3, t.b3, acc);
    return acc;
  };
  // C layout: VGPR r holds row M = r + 8*hi, col N = lane&15 of this tile.
  // No i==j test: self-sim (~1.0, the max) can only win the min for rows
  // with no other positive, which are invalid rows regardless.
  auto update = [&](const v8f& acc, int tj) {
    #pragma unroll
    for (int r = 0; r < 8; ++r) {
      float s    = acc[r];
      bool  same = (tj == ti[r]);
      float pcand = same ? s : PINF;     // v_cndmask
      float ncand = same ? -PINF : s;    // v_cndmask
      rmin[r] = vminf(rmin[r], pcand);
      rmax[r] = vmaxf(rmax[r], ncand);
    }
  };

  // Ping-pong double buffer, unroll-by-2: prefetch loads overlap the other
  // buffer's WMMAs; buffers keep stable register homes (no rotation movs).
  int ct0 = chunk * TILES_PER_CHUNK, ctEnd = ct0 + TILES_PER_CHUNK;
  BTile P = load_tile(Xh, targets, ct0, nsub, hi);
  for (int ct = ct0; ct < ctEnd; ct += 2) {
    BTile Q = load_tile(Xh, targets, ct + 1, nsub, hi);
    update(mm(P), P.tj);
    int ctn = (ct + 2 < ctEnd) ? ct + 2 : ct;   // last iter: harmless reload
    P = load_tile(Xh, targets, ctn, nsub, hi);
    update(mm(Q), Q.tj);
  }

  // reduce over the 16 N-lanes of each half-wave (masks <16 keep halves apart)
  #pragma unroll
  for (int r = 0; r < 8; ++r) {
    #pragma unroll
    for (int m = 1; m < 16; m <<= 1) {
      rmin[r] = vminf(rmin[r], __shfl_xor(rmin[r], m, 32));
      rmax[r] = vmaxf(rmax[r], __shfl_xor(rmax[r], m, 32));
    }
  }
  if (nsub == 0) {
    #pragma unroll
    for (int r = 0; r < 8; ++r) {
      int i = rowBase + r + 8 * hi;
      atomicMin(&minKey[i], enc_f32(rmin[r]));
      atomicMax(&maxKey[i], enc_f32(rmax[r]));
    }
  }
}

// ---------------------------------------------------------------------------
// Pass B: thresholded exp-sums only (no keep-counts; validity comes from the
// min/max stats in finalize). Fully branchless: both exps computed
// unconditionally (v_exp is TRANS and co-executes), masks folded via
// cndmask(0/1) + fmaf. Self exclusion kept here (it would bias pos_sum).
__global__ void __launch_bounds__(256)
pass_sums(const _Float16* __restrict__ Xh, const int* __restrict__ targets,
          const unsigned* __restrict__ minKey, const unsigned* __restrict__ maxKey,
          float* __restrict__ ps, float* __restrict__ ns) {
  int gwave = (blockIdx.x * blockDim.x + threadIdx.x) >> 5;
  int lane  = threadIdx.x & 31;
  int rowTile = gwave / COL_CHUNKS;
  int chunk   = gwave % COL_CHUNKS;
  if (rowTile >= N_TILES) return;
  int hi   = lane >> 4;
  int nsub = lane & 15;
  int rowBase = rowTile * 16;

  v16h a0 = load_frag(Xh, rowBase + nsub,  0, hi);
  v16h a1 = load_frag(Xh, rowBase + nsub, 32, hi);
  v16h a2 = load_frag(Xh, rowBase + nsub, 64, hi);
  v16h a3 = load_frag(Xh, rowBase + nsub, 96, hi);

  int ti[8]; float mp[8], mn[8];
  #pragma unroll
  for (int r = 0; r < 8; ++r) {
    int i = rowBase + r + 8 * hi;
    ti[r] = targets[i];
    mp[r] = dec_f32(minKey[i]);   // min positive sim of row i
    mn[r] = dec_f32(maxKey[i]);   // max negative sim of row i
  }

  float rpos[8] = {0,0,0,0,0,0,0,0}, rneg[8] = {0,0,0,0,0,0,0,0};

  auto mm = [&](const BTile& t) {
    v8f acc = {};
    acc = wmma_f16(a0, t.b0, acc);
    acc = wmma_f16(a1, t.b1, acc);
    acc = wmma_f16(a2, t.b2, acc);
    acc = wmma_f16(a3, t.b3, acc);
    return acc;
  };
  auto update = [&](const v8f& acc, int tj, int jr) {
    #pragma unroll
    for (int r = 0; r < 8; ++r) {
      float s    = acc[r];
      bool  same = (tj == ti[r]);
      bool  self = (jr == r);
      float pe = __expf(-K_BETA  * (s - K_MARGIN));
      float ne = __expf( K_ALPHA * (s - K_MARGIN));
      bool  posk = same && !self && (s - 0.1f < mn[r]);   // pos_keep
      bool  negk = !same          && (s + 0.1f > mp[r]);  // neg_keep
      float pm = posk ? 1.0f : 0.0f;                      // v_cndmask
      float nm = negk ? 1.0f : 0.0f;
      rpos[r] = fmaf(pm, pe, rpos[r]);
      rneg[r] = fmaf(nm, ne, rneg[r]);
    }
  };

  int ct0 = chunk * TILES_PER_CHUNK, ctEnd = ct0 + TILES_PER_CHUNK;
  int jrBase = nsub - rowBase - 8 * hi;   // jr(ct) = ct*16 + jrBase
  BTile P = load_tile(Xh, targets, ct0, nsub, hi);
  for (int ct = ct0; ct < ctEnd; ct += 2) {
    BTile Q = load_tile(Xh, targets, ct + 1, nsub, hi);
    update(mm(P), P.tj, ct * 16 + jrBase);
    int ctn = (ct + 2 < ctEnd) ? ct + 2 : ct;   // last iter: harmless reload
    P = load_tile(Xh, targets, ctn, nsub, hi);
    update(mm(Q), Q.tj, (ct + 1) * 16 + jrBase);
  }

  #pragma unroll
  for (int r = 0; r < 8; ++r) {
    #pragma unroll
    for (int m = 1; m < 16; m <<= 1) {
      rpos[r] += __shfl_xor(rpos[r], m, 32);
      rneg[r] += __shfl_xor(rneg[r], m, 32);
    }
  }
  if (nsub == 0) {
    #pragma unroll
    for (int r = 0; r < 8; ++r) {
      int i = rowBase + r + 8 * hi;
      atomicAdd(&ps[i], rpos[r]);
      atomicAdd(&ns[i], rneg[r]);
    }
  }
}

// ---------------------------------------------------------------------------
// valid(i) <=> pos_keep and neg_keep both nonempty
//          <=> min_pos - 0.1 < max_neg   (one strict compare; +inf/-inf
//              empty-set sentinels make it false automatically)
__global__ void __launch_bounds__(256)
finalize_kernel(const float* __restrict__ ps, const float* __restrict__ ns,
                const unsigned* __restrict__ minKey,
                const unsigned* __restrict__ maxKey,
                float* __restrict__ out) {
  __shared__ float sdata[8];
  float acc = 0.f;
  for (int i = blockIdx.x * blockDim.x + threadIdx.x; i < N_ROWS;
       i += gridDim.x * blockDim.x) {
    float mp = dec_f32(minKey[i]);
    float mn = dec_f32(maxKey[i]);
    bool valid = (mp - 0.1f < mn);
    float term = log1pf(ps[i]) / K_BETA + log1pf(ns[i]) / K_ALPHA;
    acc += valid ? term : 0.f;
  }
  #pragma unroll
  for (int m = 16; m >= 1; m >>= 1) acc += __shfl_xor(acc, m, 32);
  int wid = threadIdx.x >> 5;
  if ((threadIdx.x & 31) == 0) sdata[wid] = acc;
  __syncthreads();
  if (threadIdx.x < 8) {
    float v = sdata[threadIdx.x];
    #pragma unroll
    for (int m = 4; m >= 1; m >>= 1) v += __shfl_xor(v, m, 32);
    if (threadIdx.x == 0) atomicAdd(out, v / (float)N_ROWS);
  }
}

// ---------------------------------------------------------------------------
extern "C" void kernel_launch(void* const* d_in, const int* in_sizes, int n_in,
                              void* d_out, int out_size, void* d_ws, size_t ws_size,
                              hipStream_t stream) {
  const float* X       = (const float*)d_in[0];   // [16384,128] f32
  const int*   targets = (const int*)d_in[1];     // [16384] i32
  float*       out     = (float*)d_out;           // scalar loss

  char* ws = (char*)d_ws;
  _Float16* Xh     = (_Float16*)ws;                                 // 4 MB
  size_t off = (size_t)N_ROWS * DIM * sizeof(_Float16);
  unsigned* minKey = (unsigned*)(ws + off); off += N_ROWS * 4;
  unsigned* maxKey = (unsigned*)(ws + off); off += N_ROWS * 4;
  float* ps        = (float*)(ws + off);    off += N_ROWS * 4;
  float* ns        = (float*)(ws + off);

  init_kernel<<<(N_ROWS + 255) / 256, 256, 0, stream>>>(minKey, maxKey, ps, ns, out);
  normalize_kernel<<<N_ROWS / 8, 256, 0, stream>>>(X, Xh);          // 1 wave/row
  int waves  = N_TILES * COL_CHUNKS;                                // 8192 waves
  int blocks = waves / 8;                                           // 8 waves/block
  pass_minmax<<<blocks, 256, 0, stream>>>(Xh, targets, minKey, maxKey);
  pass_sums<<<blocks, 256, 0, stream>>>(Xh, targets, minKey, maxKey, ps, ns);
  finalize_kernel<<<64, 256, 0, stream>>>(ps, ns, minKey, maxKey, out);
}
